// GNNModel_24644522344815
// MI455X (gfx1250) — compile-verified, matrix operand order checked by reading, not run on hardware
//
#include <hip/hip_runtime.h>
#include <hip/hip_bf16.h>

// ---------------------------------------------------------------------------
// GNN edge-MLP message passing for MI455X (gfx1250, wave32, WMMA bf16)
//   msg = relu([x[dst] | x[src] | ea] @ W1 + b1) @ W2 + b2
//   out[dst] += mask * msg ; out = relu(out)
// ---------------------------------------------------------------------------

typedef __attribute__((ext_vector_type(16))) __bf16 v16bf;
typedef __attribute__((ext_vector_type(8)))  float  v8f;

union ABFrag {
    v16bf    v;
    unsigned u[8];
};

__device__ __forceinline__ unsigned short f2bf(float f) {
    unsigned u = __float_as_uint(f);
    u += 0x7fffu + ((u >> 16) & 1u);   // round-to-nearest-even
    return (unsigned short)(u >> 16);
}
__device__ __forceinline__ unsigned pk2(float a, float b) {
    return (unsigned)f2bf(a) | ((unsigned)f2bf(b) << 16);
}

__device__ __forceinline__ v8f wmma_bf16(const ABFrag& a, const ABFrag& b, v8f c) {
    // emits v_wmma_f32_16x16x32_bf16
    return __builtin_amdgcn_wmma_f32_16x16x32_bf16(
        /*neg_a=*/false, a.v, /*neg_b=*/false, b.v,
        /*c_mod=*/(short)0, c, /*reuse_a=*/false, /*reuse_b=*/false);
}

#define IN_CH   64
#define HID     64
#define OUT_CH  128
#define K1PAD   160            // 131 padded to 5 * 32
#define HSTRIDE 72             // 64 + 8 pad (bf16 elems), keeps dword alignment

__global__ void __launch_bounds__(256)
gnn_edge_wmma_kernel(const float* __restrict__ x,
                     const int*   __restrict__ ei,      // [2, nE]
                     const float* __restrict__ ea,      // [nE, 3]
                     const float* __restrict__ W1,      // [131, 64] row-major
                     const float* __restrict__ b1,      // [64]
                     const float* __restrict__ W2,      // [64, 128] row-major
                     const float* __restrict__ b2,      // [128]
                     float*       __restrict__ out,     // [nNodes, 128]
                     int nE)
{
    // ---- LDS ----------------------------------------------------------
    __shared__ __align__(16) unsigned short w1c[HID * K1PAD];   // 20480 B, col-major [n][k]
    __shared__ __align__(16) unsigned short w2c[OUT_CH * HID];  // 16384 B, col-major [n][k]
    __shared__ __align__(16) unsigned short htile[8][16 * HSTRIDE]; // 18432 B
    __shared__ int   srcv[8][16];
    __shared__ int   dstv[8][16];
    __shared__ float maskv[8][16];
    __shared__ float eav[8][16][4];

    const int tid = threadIdx.x;

    // ---- stage weights as bf16, column-major, K padded with zeros -----
    for (int idx = tid; idx < HID * K1PAD; idx += 256) {
        int n = idx / K1PAD, k = idx % K1PAD;
        float v = (k < 2 * IN_CH + 3) ? W1[k * HID + n] : 0.0f;
        w1c[n * K1PAD + k] = f2bf(v);
    }
    for (int idx = tid; idx < OUT_CH * HID; idx += 256) {
        int n = idx / HID, k = idx % HID;
        w2c[n * HID + k] = f2bf(W2[k * OUT_CH + n]);
    }

    const int w    = tid >> 5;
    const int lane = tid & 31;
    const int lo   = lane & 15;
    const int hi   = (lane >> 4) & 1;
    const int tile = blockIdx.x * 8 + w;
    const int e0   = tile * 16;

    // ---- per-edge metadata (rows of this wave's 16-edge tile) ---------
    if (lane < 16) {
        int e = e0 + lo;
        int s = 0, d = 0;
        float a0 = 0.f, a1 = 0.f, a2 = 0.f, mk = 0.f;
        if (e < nE) {
            s  = ei[e];
            d  = ei[nE + e];
            a0 = ea[e * 3 + 0];
            a1 = ea[e * 3 + 1];
            a2 = ea[e * 3 + 2];
            float ntype = x[(size_t)s * IN_CH];   // x_j[:,0]
            bool m = (ntype == 0.0f) ? (a0 < 0.5f)
                   : ((ntype == 1.0f) ? (a0 < 0.3f) : true);
            mk = m ? 1.0f : 0.0f;
        }
        srcv[w][lo] = s;  dstv[w][lo] = d;  maskv[w][lo] = mk;
        eav[w][lo][0] = a0; eav[w][lo][1] = a1; eav[w][lo][2] = a2;
    }
    __syncthreads();

    // ---- K-pair swizzle for 16-bit WMMA operands ----------------------
    // VGPR r of a 16-bit A/B fragment holds K = {0,2,4,6,16,18,20,22}[r]
    // (+8 for lanes 16..31), each packing (K, K+1).
    int ko[8];
#pragma unroll
    for (int r = 0; r < 8; ++r)
        ko[r] = ((r < 4) ? 2 * r : 16 + 2 * (r - 4)) + (hi ? 8 : 0);

    const int myS = srcv[w][lo];
    const int myD = dstv[w][lo];
    const float ea0 = eav[w][lo][0], ea1 = eav[w][lo][1], ea2 = eav[w][lo][2];

    // ---- build A fragments (16 edges x 160) straight from global ------
    const float* xd = x + (size_t)myD * IN_CH;   // cols   0..63  = x_i
    const float* xs = x + (size_t)myS * IN_CH;   // cols  64..127 = x_j
    ABFrag afr[5];
#pragma unroll
    for (int r = 0; r < 8; ++r) {
        afr[0].u[r] = pk2(xd[ko[r]],      xd[ko[r] + 1]);
        afr[1].u[r] = pk2(xd[32 + ko[r]], xd[32 + ko[r] + 1]);
        afr[2].u[r] = pk2(xs[ko[r]],      xs[ko[r] + 1]);
        afr[3].u[r] = pk2(xs[32 + ko[r]], xs[32 + ko[r] + 1]);
        afr[4].u[r] = 0u;
    }
    if (!hi) {                       // cols 128..130 = edge_attr, rest zero
        afr[4].u[0] = pk2(ea0, ea1);
        afr[4].u[1] = pk2(ea2, 0.0f);
    }

    // ---- GEMM1: H = relu(A @ W1 + b1)  (16x64), stage H in LDS --------
#pragma unroll
    for (int nt = 0; nt < 4; ++nt) {
        const int n1 = nt * 16 + lo;
        const float bv = b1[n1];
        v8f c;
#pragma unroll
        for (int i = 0; i < 8; ++i) c[i] = bv;

        ABFrag bf_;
#pragma unroll
        for (int kt = 0; kt < 5; ++kt) {
#pragma unroll
            for (int r = 0; r < 8; ++r)
                bf_.u[r] = *(const unsigned*)&w1c[n1 * K1PAD + kt * 32 + ko[r]];
            c = wmma_bf16(afr[kt], bf_, c);
        }
#pragma unroll
        for (int i = 0; i < 8; ++i) {
            const int M = i + hi * 8;
            htile[w][M * HSTRIDE + n1] = f2bf(fmaxf(c[i], 0.0f));
        }
    }
    __syncthreads();   // cross-lane C->A relayout through LDS

    // ---- per-output-row mask / destination ----------------------------
    float mk8[8]; int dn8[8];
#pragma unroll
    for (int i = 0; i < 8; ++i) {
        const int M = i + hi * 8;
        mk8[i] = maskv[w][M];
        dn8[i] = dstv[w][M];
    }

    // ---- GEMM2 A fragments (H as 16x64 bf16) --------------------------
    ABFrag hfr[2];
#pragma unroll
    for (int r = 0; r < 8; ++r) {
        hfr[0].u[r] = *(const unsigned*)&htile[w][lo * HSTRIDE + ko[r]];
        hfr[1].u[r] = *(const unsigned*)&htile[w][lo * HSTRIDE + 32 + ko[r]];
    }

    // ---- GEMM2: msg = H @ W2 + b2 (16x128), mask, scatter-add ---------
#pragma unroll
    for (int nt2 = 0; nt2 < 8; ++nt2) {
        const int n2 = nt2 * 16 + lo;
        const float bv = b2[n2];
        v8f c;
#pragma unroll
        for (int i = 0; i < 8; ++i) c[i] = bv;

        ABFrag bf_;
#pragma unroll
        for (int r = 0; r < 8; ++r)
            bf_.u[r] = *(const unsigned*)&w2c[n2 * HID + ko[r]];
        c = wmma_bf16(hfr[0], bf_, c);
#pragma unroll
        for (int r = 0; r < 8; ++r)
            bf_.u[r] = *(const unsigned*)&w2c[n2 * HID + 32 + ko[r]];
        c = wmma_bf16(hfr[1], bf_, c);

#pragma unroll
        for (int i = 0; i < 8; ++i) {
            atomicAdd(out + (size_t)dn8[i] * OUT_CH + n2, c[i] * mk8[i]);
        }
    }
}

__global__ void __launch_bounds__(256)
relu_inplace_kernel(float* __restrict__ p, int n4)
{
    int i = blockIdx.x * 256 + threadIdx.x;
    if (i < n4) {
        float4 v = ((float4*)p)[i];
        v.x = fmaxf(v.x, 0.0f);
        v.y = fmaxf(v.y, 0.0f);
        v.z = fmaxf(v.z, 0.0f);
        v.w = fmaxf(v.w, 0.0f);
        ((float4*)p)[i] = v;
    }
}

extern "C" void kernel_launch(void* const* d_in, const int* in_sizes, int n_in,
                              void* d_out, int out_size, void* d_ws, size_t ws_size,
                              hipStream_t stream)
{
    const float* x  = (const float*)d_in[0];
    const int*   ei = (const int*)  d_in[1];
    const float* ea = (const float*)d_in[2];
    const float* W1 = (const float*)d_in[3];
    const float* b1 = (const float*)d_in[4];
    const float* W2 = (const float*)d_in[5];
    const float* b2 = (const float*)d_in[6];
    float* out = (float*)d_out;

    const int nE = in_sizes[1] / 2;

    // segment_sum needs a zeroed accumulator (d_out is poisoned by harness).
    hipMemsetAsync(out, 0, (size_t)out_size * sizeof(float), stream);

    // One wave per 16 edges; 8 waves (128 edges) per 256-thread block.
    const int blocks = (nE + 127) / 128;
    gnn_edge_wmma_kernel<<<blocks, 256, 0, stream>>>(x, ei, ea, W1, b1, W2, b2,
                                                     out, nE);

    const int n4 = out_size / 4;
    relu_inplace_kernel<<<(n4 + 255) / 256, 256, 0, stream>>>(out, n4);
}
